// MoELayer_18683107737928
// MI455X (gfx1250) — compile-verified, matrix operand order checked by reading, not run on hardware
//
#include <hip/hip_runtime.h>
#include <hip/hip_bf16.h>
#include <cstdint>

#define B_      4
#define S_      2048
#define H_      1024
#define I_      1024
#define E_      32
#define K_      4
#define T_      (B_ * S_)          // 8192 tokens
#define N_      (T_ * K_)          // 32768 (token,expert) pairs
#define CAP_    ((N_ / E_) * 2)    // 2048 per-expert capacity
#define LIMIT_  7.0f
#define TM_     32                 // tokens per workgroup tile (2 WMMA M-tiles)

#define W1ELEMS ((size_t)E_ * 2 * I_ * H_)   // 67108864
#define W2ELEMS ((size_t)E_ * H_ * I_)       // 33554432

typedef __attribute__((ext_vector_type(16))) __bf16 v16bf;
typedef __attribute__((ext_vector_type(8)))  float  v8f;

union Frag {
    v16bf v;
    unsigned short s[16];
    uint4 q[2];
};

// fp32 -> bf16 (round-to-nearest-even) via bit ops
static __device__ __forceinline__ unsigned int f2bf(float f) {
    union { float f; unsigned int u; } c; c.f = f;
    unsigned int u = c.u;
    return (u + 0x7FFFu + ((u >> 16) & 1u)) >> 16;
}

// ---------------------------------------------------------------------------
// 0) zero output + counters (output accumulated with atomics; ws is poisoned)
__global__ void moe_init(float* __restrict__ y, int* __restrict__ cnt,
                         float* __restrict__ psum) {
    int i = blockIdx.x * blockDim.x + threadIdx.x;
    int stride = gridDim.x * blockDim.x;
    for (int j = i; j < T_ * H_ + 1; j += stride) y[j] = 0.0f;
    if (i < E_) { cnt[i] = 0; psum[i] = 0.0f; }
}

// ---------------------------------------------------------------------------
// 0b) convert expert weights fp32 -> bf16 once per launch (memory-bound,
//     ~25us of HBM traffic; removes all conversion VALU from the GEMM loop)
__global__ void moe_convert(const float* __restrict__ w1,
                            const float* __restrict__ w2,
                            unsigned int* __restrict__ bw1,  // packed 2x bf16
                            unsigned int* __restrict__ bw2) {
    size_t i0 = (size_t)blockIdx.x * blockDim.x + threadIdx.x;
    size_t stride = (size_t)gridDim.x * blockDim.x;
    const size_t n1 = W1ELEMS / 4;   // in float4 units
    const size_t n2 = W2ELEMS / 4;
    for (size_t i = i0; i < n1; i += stride) {
        float4 v = ((const float4*)w1)[i];
        uint2 o;
        o.x = f2bf(v.x) | (f2bf(v.y) << 16);
        o.y = f2bf(v.z) | (f2bf(v.w) << 16);
        ((uint2*)bw1)[i] = o;
    }
    for (size_t i = i0; i < n2; i += stride) {
        float4 v = ((const float4*)w2)[i];
        uint2 o;
        o.x = f2bf(v.x) | (f2bf(v.y) << 16);
        o.y = f2bf(v.z) | (f2bf(v.w) << 16);
        ((uint2*)bw2)[i] = o;
    }
}

// ---------------------------------------------------------------------------
// 1) router: one wave32 per token, lane == expert (E_ == 32)
__global__ __launch_bounds__(256) void moe_router(
    const float* __restrict__ x, const float* __restrict__ gw,
    int* __restrict__ topi, float* __restrict__ topw,
    float* __restrict__ psum) {
    const int lane  = threadIdx.x & 31;
    const int wid   = threadIdx.x >> 5;
    const int token = blockIdx.x * 8 + wid;

    const float* xr = x  + (size_t)token * H_;
    const float* gr = gw + (size_t)lane  * H_;
    float acc = 0.0f;
    #pragma unroll 4
    for (int h = 0; h < H_; h += 4) {
        float4 a = *(const float4*)(xr + h);
        float4 g = *(const float4*)(gr + h);
        acc = fmaf(a.x, g.x, acc);
        acc = fmaf(a.y, g.y, acc);
        acc = fmaf(a.z, g.z, acc);
        acc = fmaf(a.w, g.w, acc);
    }

    // full softmax over E for the aux loss
    float mx = acc;
    for (int o = 16; o; o >>= 1) mx = fmaxf(mx, __shfl_xor(mx, o, 32));
    float ex = __expf(acc - mx);
    float sm = ex;
    for (int o = 16; o; o >>= 1) sm += __shfl_xor(sm, o, 32);
    atomicAdd(&psum[lane], ex * __builtin_amdgcn_rcpf(sm));

    // iterative in-wave top-4 argmax
    float cur = acc;
    float tv[K_]; int ti[K_];
    #pragma unroll
    for (int j = 0; j < K_; ++j) {
        float mv = cur;
        for (int o = 16; o; o >>= 1) mv = fmaxf(mv, __shfl_xor(mv, o, 32));
        unsigned long long ball = __ballot(cur == mv);
        int widx = __ffsll(ball) - 1;
        tv[j] = mv; ti[j] = widx;
        if (lane == widx) cur = -3.0e38f;
    }
    // softmax over top-4 (tv sorted descending)
    float z = 0.0f;
    #pragma unroll
    for (int j = 0; j < K_; ++j) z += __expf(tv[j] - tv[0]);
    float rz = __builtin_amdgcn_rcpf(z);
    if (lane < K_) {
        topi[token * K_ + lane] = ti[lane];
        topw[token * K_ + lane] = __expf(tv[lane] - tv[0]) * rz;
    }
}

// ---------------------------------------------------------------------------
// 2) dispatch: pack (token, weight) into per-expert capacity buffers
__global__ void moe_dispatch(const int* __restrict__ topi,
                             const float* __restrict__ topw,
                             int* __restrict__ cnt,
                             int* __restrict__ slot_tok,
                             float* __restrict__ slot_w) {
    int idx = blockIdx.x * blockDim.x + threadIdx.x;   // 0..N_-1
    if (idx >= N_) return;
    int token = idx >> 2;
    int e = topi[idx];
    float w = topw[idx];
    int pos = atomicAdd(&cnt[e], 1);
    if (pos < CAP_) {
        slot_tok[e * CAP_ + pos] = token;
        slot_w[e * CAP_ + pos]   = w;
    }
}

// ---------------------------------------------------------------------------
// 3) fused expert MLP, 32-token tiles: gather -> WMMA GEMM1 -> SwiGLU ->
//    WMMA GEMM2 -> weighted atomic scatter.  Each B (weight) fragment feeds
//    TWO WMMAs (two M-tiles) -> half the weight-stream bandwidth per matrix
//    op.  K-loop is software-pipelined (prefetch next fragments before wmma).
__global__ __launch_bounds__(256) void moe_expert(
    const float* __restrict__ x,
    const unsigned short* __restrict__ bw1, const float* __restrict__ b1,
    const unsigned short* __restrict__ bw2, const float* __restrict__ b2,
    const int* __restrict__ cnt,
    const int* __restrict__ slot_tok, const float* __restrict__ slot_w,
    float* __restrict__ y) {
    const int e     = blockIdx.y;
    const int tile0 = blockIdx.x * TM_;
    int count = cnt[e]; if (count > CAP_) count = CAP_;
    if (tile0 >= count) return;

    extern __shared__ __align__(16) unsigned short smem[];
    unsigned short* xs  = smem;                // [32][1024] bf16 token rows
    unsigned short* h1s = smem + TM_ * 1024;   // [32][1024] bf16 activations

    const int tid  = threadIdx.x;
    const int lane = tid & 31;
    const int wv   = tid >> 5;        // wave 0..7
    const int half = lane >> 4;       // lane-half selects K sub-range
    const int lr   = lane & 15;

    // ---- gather 32 token rows, fp32 -> bf16 into LDS ----
    {
        int row  = tid >> 3;              // 0..31
        int col0 = (tid & 7) * 128;       // 128 elems per thread
        int s = tile0 + row;
        int tok = (s < count) ? slot_tok[e * CAP_ + s] : -1;
        unsigned int* d = (unsigned int*)(xs + row * 1024 + col0);
        if (tok >= 0) {
            const float* xr = x + (size_t)tok * H_;
            #pragma unroll
            for (int c = 0; c < 128; c += 4) {
                float4 v = *(const float4*)(xr + col0 + c);
                d[(c >> 1) + 0] = f2bf(v.x) | (f2bf(v.y) << 16);
                d[(c >> 1) + 1] = f2bf(v.z) | (f2bf(v.w) << 16);
            }
        } else {
            #pragma unroll
            for (int c = 0; c < 64; ++c) d[c] = 0;
        }
    }
    __syncthreads();

    // ---- phase 1: h1 = swiglu(x_tile @ W1[e]^T + b1[e])  (32 x 2048) ----
    const unsigned short* w1e = bw1 + (size_t)e * (2 * I_) * H_;
    const unsigned short* a0base = xs + lr * 1024 + half * 8;          // M 0-15
    const unsigned short* a1base = xs + (16 + lr) * 1024 + half * 8;   // M 16-31
    for (int nt = 0; nt < 16; ++nt) {
        const int n0   = wv * 256 + nt * 16;   // this wave's channel tile
        const int chan = n0 + lr;
        v8f acc0 = {0.f, 0.f, 0.f, 0.f, 0.f, 0.f, 0.f, 0.f};
        v8f acc1 = {0.f, 0.f, 0.f, 0.f, 0.f, 0.f, 0.f, 0.f};
        const unsigned short* wrow = w1e + (size_t)chan * H_ + half * 8;
        Frag a0c, a1c, bc;
        a0c.q[0] = *(const uint4*)(a0base);
        a0c.q[1] = *(const uint4*)(a0base + 16);
        a1c.q[0] = *(const uint4*)(a1base);
        a1c.q[1] = *(const uint4*)(a1base + 16);
        bc.q[0]  = *(const uint4*)(wrow);
        bc.q[1]  = *(const uint4*)(wrow + 16);
        #pragma unroll 2
        for (int k0 = 32; k0 < H_; k0 += 32) {
            Frag a0n, a1n, bn;                 // prefetch next K-step
            a0n.q[0] = *(const uint4*)(a0base + k0);
            a0n.q[1] = *(const uint4*)(a0base + k0 + 16);
            a1n.q[0] = *(const uint4*)(a1base + k0);
            a1n.q[1] = *(const uint4*)(a1base + k0 + 16);
            bn.q[0]  = *(const uint4*)(wrow + k0);
            bn.q[1]  = *(const uint4*)(wrow + k0 + 16);
            acc0 = __builtin_amdgcn_wmma_f32_16x16x32_bf16(
                false, a0c.v, false, bc.v, (short)0, acc0, false, false);
            acc1 = __builtin_amdgcn_wmma_f32_16x16x32_bf16(
                false, a1c.v, false, bc.v, (short)0, acc1, false, false);
            a0c = a0n; a1c = a1n; bc = bn;
        }
        acc0 = __builtin_amdgcn_wmma_f32_16x16x32_bf16(
            false, a0c.v, false, bc.v, (short)0, acc0, false, false);
        acc1 = __builtin_amdgcn_wmma_f32_16x16x32_bf16(
            false, a1c.v, false, bc.v, (short)0, acc1, false, false);

        const float bias = b1[(size_t)e * (2 * I_) + chan];
        // SwiGLU across interleaved channel pairs: even lane = glu, odd = lin
        #pragma unroll
        for (int r = 0; r < 8; ++r) {
            float v0 = acc0[r] + bias;
            float v1 = acc1[r] + bias;
            float p0 = __shfl_xor(v0, 1, 32);   // partner channel's value
            float p1 = __shfl_xor(v1, 1, 32);
            if ((lr & 1) == 0) {
                int m = r + half * 8;
                float glu0 = fminf(v0, LIMIT_);
                float lin0 = fminf(fmaxf(p0, -LIMIT_), LIMIT_);
                float sg0  = __builtin_amdgcn_rcpf(1.0f + __expf(-1.702f * glu0));
                h1s[m * 1024 + (n0 >> 1) + (lr >> 1)] =
                    (unsigned short)f2bf(glu0 * sg0 * (lin0 + 1.0f));
                float glu1 = fminf(v1, LIMIT_);
                float lin1 = fminf(fmaxf(p1, -LIMIT_), LIMIT_);
                float sg1  = __builtin_amdgcn_rcpf(1.0f + __expf(-1.702f * glu1));
                h1s[(m + 16) * 1024 + (n0 >> 1) + (lr >> 1)] =
                    (unsigned short)f2bf(glu1 * sg1 * (lin1 + 1.0f));
            }
        }
    }
    __syncthreads();

    // ---- phase 2: out = h1 @ W2[e]^T + b2[e], weight + scatter-add ----
    int   rtok[2][8];
    float rwgt[2][8];
    #pragma unroll
    for (int t = 0; t < 2; ++t)
        #pragma unroll
        for (int r = 0; r < 8; ++r) {
            int s = tile0 + t * 16 + r + half * 8;
            rtok[t][r] = (s < count) ? slot_tok[e * CAP_ + s] : -1;
            rwgt[t][r] = (s < count) ? slot_w[e * CAP_ + s] : 0.0f;
        }
    const unsigned short* w2e = bw2 + (size_t)e * H_ * I_;
    const unsigned short* h0base = h1s + lr * 1024 + half * 8;
    const unsigned short* h1base = h1s + (16 + lr) * 1024 + half * 8;
    for (int nt = 0; nt < 8; ++nt) {
        const int n0   = wv * 128 + nt * 16;   // output hidden-channel tile
        const int chan = n0 + lr;
        v8f acc0 = {0.f, 0.f, 0.f, 0.f, 0.f, 0.f, 0.f, 0.f};
        v8f acc1 = {0.f, 0.f, 0.f, 0.f, 0.f, 0.f, 0.f, 0.f};
        const unsigned short* wrow = w2e + (size_t)chan * I_ + half * 8;
        Frag a0c, a1c, bc;
        a0c.q[0] = *(const uint4*)(h0base);
        a0c.q[1] = *(const uint4*)(h0base + 16);
        a1c.q[0] = *(const uint4*)(h1base);
        a1c.q[1] = *(const uint4*)(h1base + 16);
        bc.q[0]  = *(const uint4*)(wrow);
        bc.q[1]  = *(const uint4*)(wrow + 16);
        #pragma unroll 2
        for (int k0 = 32; k0 < I_; k0 += 32) {
            Frag a0n, a1n, bn;
            a0n.q[0] = *(const uint4*)(h0base + k0);
            a0n.q[1] = *(const uint4*)(h0base + k0 + 16);
            a1n.q[0] = *(const uint4*)(h1base + k0);
            a1n.q[1] = *(const uint4*)(h1base + k0 + 16);
            bn.q[0]  = *(const uint4*)(wrow + k0);
            bn.q[1]  = *(const uint4*)(wrow + k0 + 16);
            acc0 = __builtin_amdgcn_wmma_f32_16x16x32_bf16(
                false, a0c.v, false, bc.v, (short)0, acc0, false, false);
            acc1 = __builtin_amdgcn_wmma_f32_16x16x32_bf16(
                false, a1c.v, false, bc.v, (short)0, acc1, false, false);
            a0c = a0n; a1c = a1n; bc = bn;
        }
        acc0 = __builtin_amdgcn_wmma_f32_16x16x32_bf16(
            false, a0c.v, false, bc.v, (short)0, acc0, false, false);
        acc1 = __builtin_amdgcn_wmma_f32_16x16x32_bf16(
            false, a1c.v, false, bc.v, (short)0, acc1, false, false);

        const float bias = b2[(size_t)e * H_ + chan];
        #pragma unroll
        for (int r = 0; r < 8; ++r) {
            if (rtok[0][r] >= 0)
                atomicAdd(y + (size_t)rtok[0][r] * H_ + chan,
                          (acc0[r] + bias) * rwgt[0][r]);
            if (rtok[1][r] >= 0)
                atomicAdd(y + (size_t)rtok[1][r] * H_ + chan,
                          (acc1[r] + bias) * rwgt[1][r]);
        }
    }
}

// ---------------------------------------------------------------------------
// 4) aux loss: one wave, lane == expert
__global__ void moe_aux(const int* __restrict__ cnt,
                        const float* __restrict__ psum,
                        float* __restrict__ y) {
    int lane = threadIdx.x;
    float v = ((float)cnt[lane] / (float)T_) * (psum[lane] / (float)T_);
    for (int o = 16; o; o >>= 1) v += __shfl_xor(v, o, 32);
    if (lane == 0) y[(size_t)T_ * H_] = 0.01f * (float)E_ * v;
}

// ---------------------------------------------------------------------------
extern "C" void kernel_launch(void* const* d_in, const int* in_sizes, int n_in,
                              void* d_out, int out_size, void* d_ws, size_t ws_size,
                              hipStream_t stream) {
    const float* x  = (const float*)d_in[0];   // (B,S,H)
    const float* gw = (const float*)d_in[1];   // (E,H)
    const float* w1 = (const float*)d_in[2];   // (E,2I,H)
    const float* b1 = (const float*)d_in[3];   // (E,2I)
    const float* w2 = (const float*)d_in[4];   // (E,H,I)
    const float* b2 = (const float*)d_in[5];   // (E,H)
    float* y = (float*)d_out;                  // (B,S,H) then aux scalar

    // ws layout: bf16 weight copies first (16B alignment for fragments)
    unsigned short* bw1 = (unsigned short*)d_ws;     // W1ELEMS bf16 (128 MB)
    unsigned short* bw2 = bw1 + W1ELEMS;             // W2ELEMS bf16 ( 64 MB)
    int*   topi = (int*)(bw2 + W2ELEMS);             // N_
    float* topw = (float*)(topi + N_);               // N_
    int*   cnt  = (int*)(topw + N_);                 // E_
    float* psum = (float*)(cnt + E_);                // E_
    int*   stok = (int*)(psum + E_);                 // E_*CAP_
    float* sw   = (float*)(stok + E_ * CAP_);        // E_*CAP_

    moe_init<<<2048, 256, 0, stream>>>(y, cnt, psum);
    moe_convert<<<4096, 256, 0, stream>>>(w1, w2,
        (unsigned int*)bw1, (unsigned int*)bw2);
    moe_router<<<T_ / 8, 256, 0, stream>>>(x, gw, topi, topw, psum);
    moe_dispatch<<<N_ / 256, 256, 0, stream>>>(topi, topw, cnt, stok, sw);
    size_t smem = (size_t)2 * TM_ * 1024 * sizeof(unsigned short);  // 128 KB
    moe_expert<<<dim3(CAP_ / TM_, E_), 256, smem, stream>>>(
        x, bw1, b1, bw2, b2, cnt, stok, sw, y);
    moe_aux<<<1, 32, 0, stream>>>(cnt, psum, y);
}
